// TTTLayer_54022098649263
// MI455X (gfx1250) — matrix operand order repaired and checked by reading, not compiled, not run on hardware
//
#include <hip/hip_runtime.h>
#include <hip/hip_bf16.h>

// ---------------------------------------------------------------------------
// Problem constants (from reference)
// ---------------------------------------------------------------------------
constexpr int kB   = 4;
constexpr int kS   = 4096;
constexpr int kHID = 768;
constexpr int kTT  = 2048;   // TTT_H
constexpr int kCS  = 512;
constexpr int kNC  = 8;
constexpr int kRC  = kB * kCS;       // 2048 rows per chunk (batch folded in)
constexpr float kLR = 0.001f;

typedef __attribute__((ext_vector_type(16))) __bf16 v16bf;
typedef __attribute__((ext_vector_type(8)))  float  v8f;

static __device__ __forceinline__ __bf16 tobf(float x) { return (__bf16)x; }

union Frag { v16bf v; uint4 q[2]; };

// gfx1250 async global->LDS copy (bypasses VGPRs, tracked by ASYNCcnt).
// INST_OFFSET is added to BOTH the LDS and global addresses (ISA 08 §4.4).
#define ASYNC_B128(ldsaddr, gaddr, OFF)                                        \
    asm volatile("global_load_async_to_lds_b128 %0, %1, off offset:" #OFF      \
                 :: "v"(ldsaddr), "v"(gaddr) : "memory")

static __device__ __forceinline__ void wait_async0() {
    asm volatile("s_wait_asynccnt 0x0" ::: "memory");
}

// ---------------------------------------------------------------------------
// Generic TN GEMM:  C[M,N] (f32) = A[M,K] (bf16 row-major, lda)
//                               @ Bt[N,K]^T (bf16 row-major, ldb)
// Block tile 128x128x64, 8 wave32 waves (4 along M x 2 along N),
// wave tile 32x64 = 2x4 WMMA tiles, two K-halves per staged tile:
// 16 v_wmma per iteration, 64 accumulator VGPRs.
// Double-buffered dynamic LDS; next tile staged with async-to-LDS copies
// while current tile's WMMAs run; ONE barrier per iteration.
// Requires M%128==0, N%128==0, K%64==0 (true for all shapes here).
// ---------------------------------------------------------------------------
__global__ __launch_bounds__(256) void gemm_tn(const __bf16* __restrict__ A, int lda,
                                               const __bf16* __restrict__ Bt, int ldb,
                                               float* __restrict__ C, int ldc, int K)
{
    constexpr int PK   = 72;                    // padded K pitch (elements), 144B rows
    constexpr int TILE = 128 * PK;              // one operand tile (elements)
    extern __shared__ char smem_raw[];          // 4*TILE*2 = 73728 bytes (dynamic)
    __bf16* smem = reinterpret_cast<__bf16*>(smem_raw);

    const int tid  = threadIdx.x;
    const int lane = tid & 31;
    const int wave = tid >> 5;
    const int wm   = wave & 3;                  // 4 waves along M
    const int wn   = wave >> 2;                 // 2 waves along N
    const int m0   = blockIdx.y * 128;
    const int n0   = blockIdx.x * 128;

    const int srow  = tid >> 1;                 // staging: 2 threads per row
    const int shalf = (tid & 1) * 32;           // elements 0..31 or 32..63

    const int lm = lane & 15;
    const int lk = lane >> 4;                   // 0 or 1 (K half-lane select)

    v8f acc[2][4] = {};

    const __bf16* gA = A  + (size_t)(m0 + srow) * lda + shalf;
    const __bf16* gB = Bt + (size_t)(n0 + srow) * ldb + shalf;

    // per-thread async staging of one 128x64 A tile + B tile into buffer `d`
    auto stage = [&](int d, int kk) {
        __bf16* Ad = smem + d * 2 * TILE;
        __bf16* Bd = Ad + TILE;
        unsigned lA = (unsigned)(size_t)(Ad + srow * PK + shalf);
        unsigned lB = (unsigned)(size_t)(Bd + srow * PK + shalf);
        const void* pA = (const void*)(gA + kk);
        const void* pB = (const void*)(gB + kk);
        ASYNC_B128(lA, pA, 0);  ASYNC_B128(lA, pA, 16);
        ASYNC_B128(lA, pA, 32); ASYNC_B128(lA, pA, 48);
        ASYNC_B128(lB, pB, 0);  ASYNC_B128(lB, pB, 16);
        ASYNC_B128(lB, pB, 32); ASYNC_B128(lB, pB, 48);
    };

    // prologue: stage tile 0 into buffer 0
    stage(0, 0);
    wait_async0();
    __syncthreads();

    int buf = 0;
    for (int k0 = 0; k0 < K; k0 += 64, buf ^= 1) {
        // stage NEXT tile into the other buffer while this tile computes
        if (k0 + 64 < K) {
            stage(buf ^ 1, k0 + 64);
            if (k0 + 128 < K) {                 // warm L2 one more tile out
                __builtin_prefetch(gA + k0 + 128, 0, 1);
                __builtin_prefetch(gB + k0 + 128, 0, 1);
            }
        }

        const __bf16* As = smem + buf * 2 * TILE;
        const __bf16* Bs = As + TILE;

        // -------- two WMMA K-steps on the current tile --------
        #pragma unroll
        for (int kh = 0; kh < 2; ++kh) {
            Frag fa[2];
            #pragma unroll
            for (int i = 0; i < 2; ++i) {
                const int m = wm * 32 + i * 16 + lm;
                fa[i].q[0] = *reinterpret_cast<const uint4*>(&As[m * PK + kh * 32 + lk * 8]);
                fa[i].q[1] = *reinterpret_cast<const uint4*>(&As[m * PK + kh * 32 + 16 + lk * 8]);
            }
            #pragma unroll
            for (int j = 0; j < 4; ++j) {
                Frag fb;
                const int n = wn * 64 + j * 16 + lm;
                fb.q[0] = *reinterpret_cast<const uint4*>(&Bs[n * PK + kh * 32 + lk * 8]);
                fb.q[1] = *reinterpret_cast<const uint4*>(&Bs[n * PK + kh * 32 + 16 + lk * 8]);
                #pragma unroll
                for (int i = 0; i < 2; ++i)
                    acc[i][j] = __builtin_amdgcn_wmma_f32_16x16x32_bf16(
                        false, fa[i].v, false, fb.v, (short)0, acc[i][j], false, false);
            }
        }

        wait_async0();                          // our async writes to buf^1 landed
        __syncthreads();                        // everyone done reading buf / writing buf^1
    }

    // -------- epilogue: f32 C tiles, documented 16x16 layout --------
    #pragma unroll
    for (int i = 0; i < 2; ++i)
        #pragma unroll
        for (int j = 0; j < 4; ++j)
            #pragma unroll
            for (int r = 0; r < 8; ++r) {
                const int row = m0 + wm * 32 + i * 16 + lk * 8 + r;
                const int col = n0 + wn * 64 + j * 16 + lm;
                C[(size_t)row * ldc + col] = acc[i][j][r];
            }
}

constexpr int kGemmLds = 4 * 128 * 72 * 2;      // 73,728 bytes dynamic LDS

// ---------------------------------------------------------------------------
// Elementwise / transpose / reduction kernels
// ---------------------------------------------------------------------------
__global__ void conv_n_k(const float* __restrict__ in, __bf16* __restrict__ out, int n) {
    int i = blockIdx.x * 256 + threadIdx.x;
    if (i < n) out[i] = tobf(in[i]);
}

// in: f32 [R][Cc] row-major  ->  out: bf16 [Cc][R]
__global__ void conv_t_k(const float* __restrict__ in, __bf16* __restrict__ out, int R, int Cc) {
    int i = blockIdx.x * 256 + threadIdx.x;
    if (i >= R * Cc) return;
    int r = i / Cc, c = i - r * Cc;
    out[(size_t)c * R + r] = tobf(in[i]);
}

// qkv f32 [B*S, 2304] -> kb bf16 [NC*RC,768] (chunk-major), kT bf16 [NC,768,RC], vf f32 [NC*RC,768]
__global__ void split_k(const float* __restrict__ qkv, __bf16* __restrict__ kb,
                        __bf16* __restrict__ kT, float* __restrict__ vf) {
    int i = blockIdx.x * 256 + threadIdx.x;            // over B*S*768
    if (i >= kB * kS * kHID) return;
    int g = i / kHID, f = i - g * kHID;
    int b = g >> 12, s = g & 4095;                     // S = 4096
    int c = s >> 9,  sl = s & 511;                     // CS = 512
    int r  = b * kCS + sl;                             // row within chunk
    int cr = c * kRC + r;                              // chunk-major row
    float kv = qkv[(size_t)g * 2304 + 768 + f];
    float vv = qkv[(size_t)g * 2304 + 1536 + f];
    kb[(size_t)cr * kHID + f] = tobf(kv);
    kT[(size_t)c * kHID * kRC + (size_t)f * kRC + r] = tobf(kv);
    vf[(size_t)cr * kHID + f] = vv;
}

// gate = silu(Gpre); GH = gate*H -> bf16 (and transposed copy for g1 GEMM)
__global__ void e1_k(const float* __restrict__ Gp, const float* __restrict__ Hf,
                     __bf16* __restrict__ GHb, __bf16* __restrict__ GHT, int writeT) {
    int i = blockIdx.x * 256 + threadIdx.x;            // over RC*TTT_H
    if (i >= kRC * kTT) return;
    float g  = Gp[i], hd = Hf[i];
    float sg = 1.f / (1.f + __expf(-g));
    float gh = g * sg * hd;
    GHb[i] = tobf(gh);
    if (writeT) { int r = i >> 11, h = i & 2047; GHT[(size_t)h * kRC + r] = tobf(gh); }
}

// D = c2*(vpred - v); emit Db, D^T bf16; block-partial sum of (vpred-v)^2
__global__ void e2_k(const float* __restrict__ vp, const float* __restrict__ vch,
                     __bf16* __restrict__ Db, __bf16* __restrict__ DT,
                     float* __restrict__ lossP, float c2) {
    __shared__ float sd[256];
    int i = blockIdx.x * 256 + threadIdx.x;            // over RC*HID (exact multiple)
    float d0 = vp[i] - vch[i];
    float dv = c2 * d0;
    int r = i / kHID, n = i - r * kHID;
    Db[i] = tobf(dv);
    DT[(size_t)n * kRC + r] = tobf(dv);
    sd[threadIdx.x] = d0 * d0;
    __syncthreads();
    for (int s = 128; s > 0; s >>= 1) {
        if (threadIdx.x < s) sd[threadIdx.x] += sd[threadIdx.x + s];
        __syncthreads();
    }
    if (threadIdx.x == 0) lossP[blockIdx.x] = sd[0];
}

// backward through SwiGLU: dgate_pre^T, dhidden^T (bf16, transposed for K-contig GEMMs)
__global__ void e3_k(const float* __restrict__ dHf, const float* __restrict__ Gp,
                     const float* __restrict__ Hf, __bf16* __restrict__ DgT,
                     __bf16* __restrict__ dHidT) {
    int i = blockIdx.x * 256 + threadIdx.x;            // over RC*TTT_H
    if (i >= kRC * kTT) return;
    int r = i >> 11, h = i & 2047;
    float dh = dHf[i], g = Gp[i], hd = Hf[i];
    float sg   = 1.f / (1.f + __expf(-g));
    float gate = g * sg;
    float dgp  = dh * hd * (sg * (1.f + g * (1.f - sg)));   // d(pre-act) via silu'
    float dhid = dh * gate;
    DgT  [(size_t)h * kRC + r] = tobf(dgp);
    dHidT[(size_t)h * kRC + r] = tobf(dhid);
}

// SGD update of f32 master weight [R][Cc]; re-emit bf16 natural and/or transposed
__global__ void upd_k(float* __restrict__ w, const float* __restrict__ g,
                      __bf16* __restrict__ outN, __bf16* __restrict__ outT, int R, int Cc) {
    int i = blockIdx.x * 256 + threadIdx.x;
    if (i >= R * Cc) return;
    int r = i / Cc, c = i - r * Cc;
    float v = w[i] - kLR * g[i];
    w[i] = v;
    if (outN) outN[i] = tobf(v);
    outT[(size_t)c * R + r] = tobf(v);
}

// y = x + q * ttt_out (chunk-major -> seq order), emit bf16 for output GEMM
__global__ void combine_k(const float* __restrict__ x, const float* __restrict__ qkv,
                          const float* __restrict__ outch, __bf16* __restrict__ yb) {
    int i = blockIdx.x * 256 + threadIdx.x;            // over B*S*768
    if (i >= kB * kS * kHID) return;
    int g = i / kHID, f = i - g * kHID;
    int b = g >> 12, s = g & 4095;
    int c = s >> 9,  sl = s & 511;
    int cr = c * kRC + b * kCS + sl;
    float y = x[i] + qkv[(size_t)g * 2304 + f] * outch[(size_t)cr * kHID + f];
    yb[i] = tobf(y);
}

__global__ void lossred_k(const float* __restrict__ lossP, int n, float* __restrict__ out,
                          float scale) {
    __shared__ float sd[256];
    float a = 0.f;
    for (int i = threadIdx.x; i < n; i += 256) a += lossP[i];
    sd[threadIdx.x] = a;
    __syncthreads();
    for (int s = 128; s > 0; s >>= 1) {
        if (threadIdx.x < s) sd[threadIdx.x] += sd[threadIdx.x + s];
        __syncthreads();
    }
    if (threadIdx.x == 0) *out = sd[0] * scale;
}

// ---------------------------------------------------------------------------
// Host orchestration
// ---------------------------------------------------------------------------
extern "C" void kernel_launch(void* const* d_in, const int* in_sizes, int n_in,
                              void* d_out, int out_size, void* d_ws, size_t ws_size,
                              hipStream_t stream) {
    const float* x     = (const float*)d_in[0];   // [4,4096,768]
    const float* wqkv  = (const float*)d_in[1];   // [768,2304]
    const float* w0    = (const float*)d_in[2];   // [768,2048]
    const float* w1    = (const float*)d_in[3];   // [2048,768]
    const float* w2    = (const float*)d_in[4];   // [768,2048]
    const float* wout  = (const float*)d_in[5];   // [768,768]

    char* base = (char*)d_ws;
    size_t off = 0;
    auto alloc = [&](size_t bytes) -> void* {
        void* p = base + off;
        off += (bytes + 255) & ~(size_t)255;
        return p;
    };

    const int NTOK = kB * kS;                  // 16384
    const int NE   = NTOK * kHID;              // 12,582,912
    const int NCH  = kRC * kTT;                // 4,194,304
    const int NCHN = kRC * kHID;               // 1,572,864
    const int E2B  = NCHN / 256;               // 6144 partial blocks / chunk

    __bf16* xb     = (__bf16*)alloc((size_t)NE * 2);
    __bf16* wqkvT  = (__bf16*)alloc((size_t)2304 * 768 * 2);
    __bf16* woutT  = (__bf16*)alloc((size_t)768 * 768 * 2);
    float*  qkv    = (float*) alloc((size_t)NTOK * 2304 * 4);
    __bf16* kb     = (__bf16*)alloc((size_t)kNC * NCHN * 2);
    __bf16* kT     = (__bf16*)alloc((size_t)kNC * NCHN * 2);
    float*  vf     = (float*) alloc((size_t)kNC * NCHN * 4);
    float*  w0c    = (float*) alloc((size_t)768 * 2048 * 4);
    float*  w1c    = (float*) alloc((size_t)2048 * 768 * 4);
    float*  w2c    = (float*) alloc((size_t)768 * 2048 * 4);
    __bf16* w0T    = (__bf16*)alloc((size_t)2048 * 768 * 2);
    __bf16* w1b    = (__bf16*)alloc((size_t)2048 * 768 * 2);
    __bf16* w1T    = (__bf16*)alloc((size_t)768 * 2048 * 2);
    __bf16* w2T    = (__bf16*)alloc((size_t)2048 * 768 * 2);
    float*  Gpre   = (float*) alloc((size_t)NCH * 4);
    float*  Hf     = (float*) alloc((size_t)NCH * 4);
    float*  dHf    = (float*) alloc((size_t)NCH * 4);
    __bf16* GHb    = (__bf16*)alloc((size_t)NCH * 2);
    __bf16* GHT    = (__bf16*)alloc((size_t)NCH * 2);
    __bf16* DgT    = (__bf16*)alloc((size_t)NCH * 2);
    __bf16* dHidT  = (__bf16*)alloc((size_t)NCH * 2);
    float*  vpred  = (float*) alloc((size_t)NCHN * 4);
    __bf16* Db     = (__bf16*)alloc((size_t)NCHN * 2);
    __bf16* DT     = (__bf16*)alloc((size_t)NCHN * 2);
    float*  g0     = (float*) alloc((size_t)768 * 2048 * 4);
    float*  g1     = (float*) alloc((size_t)2048 * 768 * 4);
    float*  g2     = (float*) alloc((size_t)768 * 2048 * 4);
    float*  outch  = (float*) alloc((size_t)NE * 4);
    __bf16* yb     = (__bf16*)alloc((size_t)NE * 2);
    float*  lossP  = (float*) alloc((size_t)kNC * E2B * 4);
    (void)ws_size; (void)in_sizes; (void)n_in;

    auto gemm = [&](const __bf16* A, int lda, const __bf16* Bt, int ldb,
                    float* C, int ldc, int M, int N, int K) {
        dim3 grid(N / 128, M / 128);
        gemm_tn<<<grid, 256, kGemmLds, stream>>>(A, lda, Bt, ldb, C, ldc, K);
    };
    auto ew = [&](int n) { return dim3((n + 255) / 256); };

    // ---- one-time conversions / master-weight init ----
    conv_n_k<<<ew(NE), 256, 0, stream>>>(x, xb, NE);
    conv_t_k<<<ew(768 * 2304), 256, 0, stream>>>(wqkv, wqkvT, 768, 2304);
    conv_t_k<<<ew(768 * 768),  256, 0, stream>>>(wout, woutT, 768, 768);
    hipMemcpyAsync(w0c, w0, (size_t)768 * 2048 * 4, hipMemcpyDeviceToDevice, stream);
    hipMemcpyAsync(w1c, w1, (size_t)2048 * 768 * 4, hipMemcpyDeviceToDevice, stream);
    hipMemcpyAsync(w2c, w2, (size_t)768 * 2048 * 4, hipMemcpyDeviceToDevice, stream);
    conv_t_k<<<ew(768 * 2048), 256, 0, stream>>>(w0, w0T, 768, 2048);
    conv_t_k<<<ew(768 * 2048), 256, 0, stream>>>(w2, w2T, 768, 2048);
    conv_t_k<<<ew(2048 * 768), 256, 0, stream>>>(w1, w1T, 2048, 768);
    conv_n_k<<<ew(2048 * 768), 256, 0, stream>>>(w1, w1b, 2048 * 768);

    // ---- QKV projection ----
    gemm(xb, kHID, wqkvT, kHID, qkv, 2304, NTOK, 2304, kHID);
    split_k<<<ew(NE), 256, 0, stream>>>(qkv, kb, kT, vf);

    const float c2 = 2.f / (float)NCHN;        // d(mean sq)/d(vpred)

    // ---- sequential TTT chunk scan ----
    for (int c = 0; c < kNC; ++c) {
        const __bf16* kc  = kb + (size_t)c * NCHN;
        const __bf16* kcT = kT + (size_t)c * NCHN;
        const float*  vc  = vf + (size_t)c * NCHN;

        // forward (old weights)
        gemm(kc, kHID, w0T, kHID, Gpre, kTT, kRC, kTT, kHID);
        gemm(kc, kHID, w2T, kHID, Hf,   kTT, kRC, kTT, kHID);
        e1_k<<<ew(NCH), 256, 0, stream>>>(Gpre, Hf, GHb, GHT, 1);
        gemm(GHb, kTT, w1T, kTT, vpred, kHID, kRC, kHID, kTT);

        // loss grad seed + loss partials
        e2_k<<<E2B, 256, 0, stream>>>(vpred, vc, Db, DT, lossP + (size_t)c * E2B, c2);

        // backward
        gemm(Db, kHID, w1b, kHID, dHf, kTT, kRC, kTT, kHID);           // dZ = D @ w1^T
        e3_k<<<ew(NCH), 256, 0, stream>>>(dHf, Gpre, Hf, DgT, dHidT);
        gemm(GHT, kRC, DT,    kRC, g1, kHID, kTT, kHID, kRC);          // g1 = Z^T D
        gemm(kcT, kRC, DgT,   kRC, g0, kTT,  kHID, kTT, kRC);          // g0 = K^T dgate
        gemm(kcT, kRC, dHidT, kRC, g2, kTT,  kHID, kTT, kRC);          // g2 = K^T dhid

        // SGD update + re-emit bf16 weights
        upd_k<<<ew(768 * 2048), 256, 0, stream>>>(w0c, g0, nullptr, w0T, 768, 2048);
        upd_k<<<ew(2048 * 768), 256, 0, stream>>>(w1c, g1, w1b, w1T, 2048, 768);
        upd_k<<<ew(768 * 2048), 256, 0, stream>>>(w2c, g2, nullptr, w2T, 768, 2048);

        // forward with updated weights -> chunk output
        gemm(kc, kHID, w0T, kHID, Gpre, kTT, kRC, kTT, kHID);
        gemm(kc, kHID, w2T, kHID, Hf,   kTT, kRC, kTT, kHID);
        e1_k<<<ew(NCH), 256, 0, stream>>>(Gpre, Hf, GHb, GHT, 0);
        gemm(GHb, kTT, w1T, kTT, outch + (size_t)c * NCHN, kHID, kRC, kHID, kTT);
    }

    // ---- combine: y = x + q*ttt ; output = y @ w_out ----
    combine_k<<<ew(NE), 256, 0, stream>>>(x, qkv, outch, yb);
    gemm(yb, kHID, woutT, kHID, (float*)d_out, kHID, NTOK, kHID, kHID);

    // ---- loss = mean over chunks of per-chunk MSE ----
    lossred_k<<<1, 256, 0, stream>>>(lossP, kNC * E2B, (float*)d_out + (out_size - 1),
                                     1.f / ((float)NCHN * (float)kNC));
}
